// GraphCastDecoder_58007828299998
// MI455X (gfx1250) — compile-verified, matrix operand order checked by reading, not run on hardware
//
#include <hip/hip_runtime.h>
#include <hip/hip_bf16.h>

#define H128 128
#define N_MESH 40962
#define N_GRID 262144
#define N_EDGE 786432

typedef __bf16 v8bf  __attribute__((ext_vector_type(8)));
typedef __bf16 v16bf __attribute__((ext_vector_type(16)));
typedef float  v8f   __attribute__((ext_vector_type(8)));

// ---------------------------------------------------------------------------
// A-fragment load: 16-bit A 16x32 layout (ISA 7.12.2):
//   lanes 0-15 : row M = lane,     K = {0..7, 16..23}
//   lanes 16-31: row M = lane-16,  K = {8..15, 24..31}
// rowk points at X[row] + chunk*32 (bf16 elements), 16B-aligned.
// ---------------------------------------------------------------------------
__device__ __forceinline__ v16bf load_a_frag(const __bf16* rowk, int hf) {
    const v8bf lo = *(const v8bf*)(rowk + hf * 8);        // K = hf*8 + 0..7
    const v8bf hi = *(const v8bf*)(rowk + 16 + hf * 8);   // K = 16 + hf*8 + 0..7
    return __builtin_shufflevector(lo, hi, 0,1,2,3,4,5,6,7,8,9,10,11,12,13,14,15);
}

// ---------------------------------------------------------------------------
// Pack a row-major fp32 weight matrix W[K][128] into per-lane B fragments:
// fragment (chunk, ntile) -> 32 lanes x 16 bf16, contiguous 32B per lane.
// B 32x16 layout: lane = column (mod 16), K split across lane halves.
// ---------------------------------------------------------------------------
__device__ __forceinline__ void pack_weights(const float* __restrict__ W, __bf16* Wp,
                                             int chunks, int tid, int nthr) {
    for (int j = tid; j < chunks * 256; j += nthr) {
        const int chunk = j >> 8;
        const int rem   = j & 255;
        const int nt    = rem >> 5;
        const int lane  = rem & 31;
        const int hf    = lane >> 4;
        const int c     = lane & 15;
        v16bf w;
#pragma unroll
        for (int e = 0; e < 16; ++e) {
            const int k = chunk * 32 + ((e & 8) << 1) + hf * 8 + (e & 7);
            w[e] = (__bf16)W[k * 128 + nt * 16 + c];
        }
        *(v16bf*)(Wp + (size_t)j * 16) = w;
    }
}

// One output tile row-set: 16 rows x 128 cols per wave, KC K-chunks.
// K loop kept un-unrolled to bound the live-register window (no spills).
template <int KC>
__device__ __forceinline__ void mma_rows(const __bf16* arow,
                                         const __bf16* Wp, int lane, int hf, v8f acc[8]) {
#pragma unroll 1
    for (int kc = 0; kc < KC; ++kc) {
        const v16bf a = load_a_frag(arow + kc * 32, hf);
#pragma unroll
        for (int n = 0; n < 8; ++n) {
            const v16bf b = *(const v16bf*)(Wp + (size_t)(((kc * 8 + n) * 32 + lane) * 16));
            acc[n] = __builtin_amdgcn_wmma_f32_16x16x32_bf16(
                false, a, false, b, (short)0, acc[n], false, false);
        }
    }
}

__device__ __forceinline__ float silu_f(float x) {
    return x / (1.0f + __expf(-x));
}

// ===========================================================================
// zero workspace
// ===========================================================================
__global__ void gcast_zero(float4* __restrict__ p, long n4) {
    long i = (long)blockIdx.x * blockDim.x + threadIdx.x;
    const long stride = (long)gridDim.x * blockDim.x;
    const float4 z = make_float4(0.f, 0.f, 0.f, 0.f);
    for (; i < n4; i += stride) p[i] = z;
}

// ===========================================================================
// Edge block: 128 edges per workgroup, 8 waves, full-N per wave.
// LDS layout (bytes):
//   [0,       98304) Wp1  : 12 chunks x 8 ntiles x 32 lanes x 32B
//   [98304,  131072) Wp2  :  4 chunks x ...
//   [131072, 231424) X (128 x 392 bf16)  -- aliased later by Y (128 x 132 f32)
//   [231424, 266240) Hb (128 x 136 bf16)
// ===========================================================================
#define E_XS 392   // bf16 elements per X row (384 + 8 pad)
#define HS   136   // bf16 elements per hidden row (128 + 8 pad)
#define YS   132   // f32 elements per Y row (128 + 4 pad)

__global__ __launch_bounds__(256, 1)
void gcast_edge(const float* __restrict__ edge_f,
                const float* __restrict__ grid_f,
                const float* __restrict__ mesh_f,
                const int*   __restrict__ src_idx,
                const int*   __restrict__ dst_idx,
                const float* __restrict__ eW1, const float* __restrict__ eb1,
                const float* __restrict__ eW2, const float* __restrict__ eb2,
                const float* __restrict__ eg,  const float* __restrict__ ebeta,
                float* __restrict__ agg) {
    __shared__ __align__(16) unsigned char smem[266240];
    __bf16* Wp1 = (__bf16*)(smem);
    __bf16* Wp2 = (__bf16*)(smem + 98304);
    __bf16* X   = (__bf16*)(smem + 131072);
    float*  Y   = (float*) (smem + 131072);
    __bf16* Hb  = (__bf16*)(smem + 231424);

    const int t  = threadIdx.x;
    const int e0 = blockIdx.x * 128;

    pack_weights(eW1, Wp1, 12, t, 256);
    pack_weights(eW2, Wp2,  4, t, 256);

    // gather + concat + cvt: X[r] = [edge | mesh[src] | grid[dst]] as bf16
    for (int i = t; i < 128 * 384; i += 256) {
        const int r = i / 384, c = i - r * 384;
        const int e = e0 + r;
        float v;
        if (c < 128)       v = edge_f[(size_t)e * 128 + c];
        else if (c < 256)  v = mesh_f[(size_t)src_idx[e] * 128 + (c - 128)];
        else               v = grid_f[(size_t)dst_idx[e] * 128 + (c - 256)];
        X[r * E_XS + c] = (__bf16)v;
    }
    __syncthreads();

    const int lane = t & 31, wid = t >> 5;
    const int hf = lane >> 4, col = lane & 15;
    const int m0 = wid * 16;
    const v8f zero8 = {0.f, 0.f, 0.f, 0.f, 0.f, 0.f, 0.f, 0.f};

    v8f acc[8];
#pragma unroll
    for (int n = 0; n < 8; ++n) acc[n] = zero8;

    // layer 1: [128x384] @ [384x128]
    mma_rows<12>(X + (m0 + col) * E_XS, Wp1, lane, hf, acc);

    // SiLU epilogue -> hidden (bf16)
#pragma unroll
    for (int n = 0; n < 8; ++n) {
        const int cn = n * 16 + col;
        const float b1 = eb1[cn];
#pragma unroll
        for (int i = 0; i < 8; ++i) {
            Hb[(m0 + i + 8 * hf) * HS + cn] = (__bf16)silu_f(acc[n][i] + b1);
        }
    }
    __syncthreads();

#pragma unroll
    for (int n = 0; n < 8; ++n) acc[n] = zero8;

    // layer 2: [128x128] @ [128x128]
    mma_rows<4>(Hb + (m0 + col) * HS, Wp2, lane, hf, acc);

    // +b2 -> Y (fp32, aliases dead X)
#pragma unroll
    for (int n = 0; n < 8; ++n) {
        const int cn = n * 16 + col;
        const float b2 = eb2[cn];
#pragma unroll
        for (int i = 0; i < 8; ++i) Y[(m0 + i + 8 * hf) * YS + cn] = acc[n][i] + b2;
    }
    __syncthreads();

    // LayerNorm + edge residual + atomic scatter into agg[dst]
    const float4 g4  = *(const float4*)(eg    + lane * 4);
    const float4 be4 = *(const float4*)(ebeta + lane * 4);
#pragma unroll 1
    for (int rr = 0; rr < 16; ++rr) {
        const int row = m0 + rr;
        const float* yr = Y + row * YS + lane * 4;
        const float y0 = yr[0], y1 = yr[1], y2 = yr[2], y3 = yr[3];
        float s = y0 + y1 + y2 + y3;
        float q = y0 * y0 + y1 * y1 + y2 * y2 + y3 * y3;
#pragma unroll
        for (int off = 16; off > 0; off >>= 1) {
            s += __shfl_xor(s, off, 32);
            q += __shfl_xor(q, off, 32);
        }
        const float mu = s * (1.0f / 128.0f);
        const float rs = rsqrtf(q * (1.0f / 128.0f) - mu * mu + 1e-5f);
        const int e = e0 + row;
        const int d = dst_idx[e];
        const float* er = edge_f + (size_t)e * 128 + lane * 4;
        float* ag = agg + (size_t)d * 128 + lane * 4;
        unsafeAtomicAdd(ag + 0, (y0 - mu) * rs * g4.x + be4.x + er[0]);
        unsafeAtomicAdd(ag + 1, (y1 - mu) * rs * g4.y + be4.y + er[1]);
        unsafeAtomicAdd(ag + 2, (y2 - mu) * rs * g4.z + be4.z + er[2]);
        unsafeAtomicAdd(ag + 3, (y3 - mu) * rs * g4.w + be4.w + er[3]);
    }
}

// ===========================================================================
// Node block: 128 grid nodes per workgroup.
// LDS: Wp1n [0,65536) Wp2n [65536,98304) X/Y [98304,165888) Hb [165888,200704)
// ===========================================================================
#define N_XS 264   // 256 + 8 pad

__global__ __launch_bounds__(256, 1)
void gcast_node(const float* __restrict__ grid_f,
                const float* __restrict__ agg,
                const float* __restrict__ nW1, const float* __restrict__ nb1,
                const float* __restrict__ nW2, const float* __restrict__ nb2,
                const float* __restrict__ ng,  const float* __restrict__ nbeta,
                float* __restrict__ out) {
    __shared__ __align__(16) unsigned char smem[200704];
    __bf16* Wp1 = (__bf16*)(smem);
    __bf16* Wp2 = (__bf16*)(smem + 65536);
    __bf16* X   = (__bf16*)(smem + 98304);
    float*  Y   = (float*) (smem + 98304);
    __bf16* Hb  = (__bf16*)(smem + 165888);

    const int t  = threadIdx.x;
    const int g0 = blockIdx.x * 128;

    pack_weights(nW1, Wp1, 8, t, 256);
    pack_weights(nW2, Wp2, 4, t, 256);

    // X[r] = [grid | agg] as bf16
    for (int i = t; i < 128 * 256; i += 256) {
        const int r = i >> 8, c = i & 255;
        const size_t node = (size_t)(g0 + r);
        const float v = (c < 128) ? grid_f[node * 128 + c]
                                  : agg[node * 128 + (c - 128)];
        X[r * N_XS + c] = (__bf16)v;
    }
    __syncthreads();

    const int lane = t & 31, wid = t >> 5;
    const int hf = lane >> 4, col = lane & 15;
    const int m0 = wid * 16;
    const v8f zero8 = {0.f, 0.f, 0.f, 0.f, 0.f, 0.f, 0.f, 0.f};

    v8f acc[8];
#pragma unroll
    for (int n = 0; n < 8; ++n) acc[n] = zero8;

    mma_rows<8>(X + (m0 + col) * N_XS, Wp1, lane, hf, acc);

#pragma unroll
    for (int n = 0; n < 8; ++n) {
        const int cn = n * 16 + col;
        const float b1 = nb1[cn];
#pragma unroll
        for (int i = 0; i < 8; ++i) {
            Hb[(m0 + i + 8 * hf) * HS + cn] = (__bf16)silu_f(acc[n][i] + b1);
        }
    }
    __syncthreads();

#pragma unroll
    for (int n = 0; n < 8; ++n) acc[n] = zero8;

    mma_rows<4>(Hb + (m0 + col) * HS, Wp2, lane, hf, acc);

#pragma unroll
    for (int n = 0; n < 8; ++n) {
        const int cn = n * 16 + col;
        const float b2 = nb2[cn];
#pragma unroll
        for (int i = 0; i < 8; ++i) Y[(m0 + i + 8 * hf) * YS + cn] = acc[n][i] + b2;
    }
    __syncthreads();

    const float4 g4  = *(const float4*)(ng    + lane * 4);
    const float4 be4 = *(const float4*)(nbeta + lane * 4);
#pragma unroll 1
    for (int rr = 0; rr < 16; ++rr) {
        const int row = m0 + rr;
        const float* yr = Y + row * YS + lane * 4;
        const float y0 = yr[0], y1 = yr[1], y2 = yr[2], y3 = yr[3];
        float s = y0 + y1 + y2 + y3;
        float q = y0 * y0 + y1 * y1 + y2 * y2 + y3 * y3;
#pragma unroll
        for (int off = 16; off > 0; off >>= 1) {
            s += __shfl_xor(s, off, 32);
            q += __shfl_xor(q, off, 32);
        }
        const float mu = s * (1.0f / 128.0f);
        const float rs = rsqrtf(q * (1.0f / 128.0f) - mu * mu + 1e-5f);
        const size_t node = (size_t)(g0 + row);
        const float* gr = grid_f + node * 128 + lane * 4;
        float4 o;
        o.x = (y0 - mu) * rs * g4.x + be4.x + gr[0];
        o.y = (y1 - mu) * rs * g4.y + be4.y + gr[1];
        o.z = (y2 - mu) * rs * g4.z + be4.z + gr[2];
        o.w = (y3 - mu) * rs * g4.w + be4.w + gr[3];
        *(float4*)(out + node * 128 + lane * 4) = o;
    }
}

// ===========================================================================
extern "C" void kernel_launch(void* const* d_in, const int* in_sizes, int n_in,
                              void* d_out, int out_size, void* d_ws, size_t ws_size,
                              hipStream_t stream) {
    const float* edge_f  = (const float*)d_in[0];
    const float* grid_f  = (const float*)d_in[1];
    const float* mesh_f  = (const float*)d_in[2];
    const int*   src_idx = (const int*)  d_in[3];
    const int*   dst_idx = (const int*)  d_in[4];
    const float* eW1 = (const float*)d_in[5];
    const float* eb1 = (const float*)d_in[6];
    const float* eW2 = (const float*)d_in[7];
    const float* eb2 = (const float*)d_in[8];
    const float* eg  = (const float*)d_in[9];
    const float* ebt = (const float*)d_in[10];
    const float* nW1 = (const float*)d_in[11];
    const float* nb1 = (const float*)d_in[12];
    const float* nW2 = (const float*)d_in[13];
    const float* nb2 = (const float*)d_in[14];
    const float* ng  = (const float*)d_in[15];
    const float* nbt = (const float*)d_in[16];
    float* out = (float*)d_out;
    float* agg = (float*)d_ws;   // N_GRID * 128 f32 = 128 MiB scratch

    const long n4 = (long)N_GRID * 128 / 4;
    gcast_zero<<<2048, 256, 0, stream>>>((float4*)agg, n4);

    gcast_edge<<<N_EDGE / 128, 256, 0, stream>>>(edge_f, grid_f, mesh_f, src_idx, dst_idx,
                                                 eW1, eb1, eW2, eb2, eg, ebt, agg);

    gcast_node<<<N_GRID / 128, 256, 0, stream>>>(grid_f, agg,
                                                 nW1, nb1, nW2, nb2, ng, nbt, out);
}
